// Attention_14336600834451
// MI455X (gfx1250) — matrix operand order repaired and checked by reading, not compiled, not run on hardware
//
#include <hip/hip_runtime.h>
#include <stdint.h>

// ---------------------------------------------------------------------------
// Quadratic-state recurrence for MI455X (gfx1250, wave32, WMMA).
//
// quad[b,k] = sum_ij s_i s_j P[i,j,k]  ==  GEMM  C[32,128] = W[32,16384] x Pflat[16384,128]
// with W[b, i*128+j] = s[b,i]*s[b,j] generated on the fly from LDS-staged state.
// P (8MB fp32 -> 4MB bf16) lives in L2 (192MB); HBM traffic is negligible, so
// the kernel is per-step latency bound -> single persistent kernel, device-side
// grid barriers, bf16 WMMA (v_wmma_f32_16x16x32_bf16) with fp32 accumulation.
// Inner loops are software-pipelined: B fragment loaded 2 iterations ahead,
// A fragment (v_perm_b32-packed bf16) built 1 iteration ahead, so the
// s_wait_loadcnt before each WMMA covers a load already ~2 iterations in flight.
// ---------------------------------------------------------------------------

typedef __attribute__((ext_vector_type(16))) __bf16          v16bf;
typedef __attribute__((ext_vector_type(16))) unsigned short  v16us;
typedef __attribute__((ext_vector_type(8)))  unsigned int    v8ui;
typedef __attribute__((ext_vector_type(8)))  float           v8f;

#define D_   128
#define I_   256
#define O_   256
#define B_   32
#define S_   4096
#define EPS_ 1e-5f

#define NWG  36
#define NTHR 256

// workspace layout (bytes) -- ~4.75 MB total.  NOTE: regions are deliberately
// adjacent; pipelined tail over-reads (never consumed) land in the next region.
#define PB_OFF   ((size_t)0)                     // P packed bf16: 512 kt x 8 nt x 1KB = 4 MB
#define QB_OFF   ((size_t)4194304)               // Q packed: 8 x 8  x 1KB = 64 KB
#define RB_OFF   (QB_OFF + 65536)                // R packed: 4 x 16 x 1KB = 64 KB
#define SB_OFF   (RB_OFF + 65536)                // S packed: 8 x 16 x 1KB = 128 KB
#define HB_OFF   (SB_OFF + 131072)               // h double buffer: 2 x 32 x 128 f32 = 32 KB
#define PART_OFF (HB_OFF + 32768)                // quad partials: 16 tiles x 16 segs x 256 f32 = 256 KB
#define CNT_OFF  (PART_OFF + 262144)             // grid barrier counter

__device__ __forceinline__ unsigned short f2bf(float f) {
  union { float f; unsigned u; } v; v.f = f;
  return (unsigned short)(v.u >> 16);            // truncate; accumulate is fp32
}

// pack two fp32 -> bf16x2 in one VGPR: byte-select {f1.b3,f1.b2,f0.b3,f0.b2}
__device__ __forceinline__ unsigned pk_bf16x2(float f0, float f1) {
  return __builtin_amdgcn_perm(__builtin_bit_cast(unsigned, f1),
                               __builtin_bit_cast(unsigned, f0),
                               0x07060302u);
}

__device__ __forceinline__ v8f wmma_bf16(v16bf a, v16bf b, v8f c) {
  return __builtin_amdgcn_wmma_f32_16x16x32_bf16(false, a, false, b, (short)0, c, false, false);
}

// Build a 16x32 bf16 A-fragment row slice for this lane from an fp32 row.
// A layout (matches packed B): K = 16*(lane>=16) + e; pairs (2p,2p+1)
// are K-consecutive, so each VGPR is one v_perm of two scaled fp32 values.
__device__ __forceinline__ v16bf buildA(const float* __restrict__ row, float scale,
                                        int kbase, int laneHi8) {
  v8ui u;
#pragma unroll
  for (int p = 0; p < 8; ++p) {
    int e0 = 2 * p;
    int kk = kbase + 2 * laneHi8 + e0;
    u[p] = pk_bf16x2(scale * row[kk], scale * row[kk + 1]);
  }
  return __builtin_bit_cast(v16bf, u);
}

__device__ __forceinline__ void grid_sync(unsigned* cnt, unsigned target) {
  __syncthreads();
  if (threadIdx.x == 0) {
    __threadfence();
    __hip_atomic_fetch_add(cnt, 1u, __ATOMIC_RELEASE, __HIP_MEMORY_SCOPE_AGENT);
    while (__hip_atomic_load(cnt, __ATOMIC_ACQUIRE, __HIP_MEMORY_SCOPE_AGENT) < target) {
      __builtin_amdgcn_s_sleep(2);
    }
  }
  __syncthreads();
}

// ---------------------------------------------------------------------------
// Pack an fp32 weight into bf16 wave32 B-fragments.
//   lane<16 : N = nt*16 + lane,    K = kt*32 + e        (e = 0..15)
//   lane>=16: N = nt*16 + lane-16, K = kt*32 + 16 + e
// frag index = kt*NT + nt ; per-frag 32 lanes x 16 bf16 = 1 KB, lane-contiguous.
// src_kn=1: src is [K,N] flat (P).  src_kn=0: src is [N,K] flat (Q,R,S -> A x src^T).
// ---------------------------------------------------------------------------
__global__ void pack_wt(const float* __restrict__ src, unsigned short* __restrict__ dst,
                        int N, int K, int src_kn) {
  int gid = blockIdx.x * blockDim.x + threadIdx.x;
  int NT = N >> 4;
  int total = (K >> 5) * NT * 32;
  if (gid >= total) return;
  int lane = gid & 31;
  int frag = gid >> 5;
  int nt = frag % NT;
  int kt = frag / NT;
  int ng = nt * 16 + (lane & 15);
  int kbase = kt * 32 + ((lane & 16) ? 16 : 0);
  v16us u;
#pragma unroll
  for (int e = 0; e < 16; ++e) {
    int kg = kbase + e;
    size_t idx = src_kn ? ((size_t)kg * N + ng) : ((size_t)ng * K + kg);
    u[e] = f2bf(src[idx]);
  }
  *(v16us*)(dst + (size_t)gid * 16) = u;
}

// ---------------------------------------------------------------------------
// Persistent recurrence kernel. 36 WGs x 256 threads (8 waves/WG).
//   WGs 0..31 : quad GEMM. wave job: tile = (wg&1)*8 + waveId, seg = wg>>1.
//               seg-0 waves (WGs 0,1) also fold in x_t @ Q^T.
//   WGs 32..35: out(t-1) = h(t-1) @ R^T + x_{t-1} @ S^T  (32 tiles of 16x16).
//   WGs 0,1   : phase B = reduce 16 K-partials + LayerNorm -> h(t) (double buf).
// Two grid barriers per step.
// ---------------------------------------------------------------------------
__global__ void __launch_bounds__(NTHR)
rqs_persistent(const float* __restrict__ xg, const float* __restrict__ gamma,
               const float* __restrict__ beta, float* __restrict__ outg,
               char* __restrict__ ws) {
  __shared__ float hstage[32][132];   // padded: avoid 64-bank conflicts on row-strided reads
  __shared__ float xstage[32][264];
  __shared__ float red[2][16][16];

  const unsigned short* Pb = (const unsigned short*)(ws + PB_OFF);
  const unsigned short* Qb = (const unsigned short*)(ws + QB_OFF);
  const unsigned short* Rb = (const unsigned short*)(ws + RB_OFF);
  const unsigned short* Sb = (const unsigned short*)(ws + SB_OFF);
  float*    hbuf = (float*)(ws + HB_OFF);
  float*    part = (float*)(ws + PART_OFF);
  unsigned* cnt  = (unsigned*)(ws + CNT_OFF);

  const int wg     = blockIdx.x;
  const int tid    = threadIdx.x;
  const int wl     = tid >> 5;
  const int lane   = tid & 31;
  const int laneLo = lane & 15;
  const int laneHi = (lane & 16) >> 1;   // 0 or 8
  unsigned epoch = 0;

  for (int t = 0; t <= S_; ++t) {
    // ---- stage state / inputs into LDS ----
    bool need_h = (t >= 1) && (wg >= 32 || t < S_);
    bool need_x = (wg < 2 && t < S_) || (wg >= 32 && t >= 1);
    if (need_h) {
      const float* hsrc = hbuf + ((t - 1) & 1) * (32 * 128);
      for (int i = tid; i < 32 * 128; i += NTHR)
        hstage[i >> 7][i & 127] = hsrc[i];
    }
    if (need_x) {
      int xt = (wg >= 32) ? (t - 1) : t;
      for (int i = tid; i < 32 * 256; i += NTHR) {
        int b = i >> 8, ii = i & 255;
        xstage[b][ii] = xg[((size_t)b * S_ + xt) * 256 + ii];
      }
    }
    __syncthreads();

    // ---- phase A ----
    if (wg < 32) {
      if (t < S_) {
        const int tile = (wg & 1) * 8 + wl;     // 0..15
        const int seg  = wg >> 1;               // 0..15 (K segment of 1024)
        const int m = tile >> 3, n = tile & 7;
        const int b = m * 16 + laneLo;
        const float* hrow = &hstage[b][0];
        v8f c = {0.f, 0.f, 0.f, 0.f, 0.f, 0.f, 0.f, 0.f};
        if (t > 0) {
          const int ktBase = seg * 32;
          auto loadB = [&](int st) -> v16bf {
            return *(const v16bf*)(Pb + (((size_t)(ktBase + st) * 8 + n) * 32 + lane) * 16);
          };
          auto mkA = [&](int st) -> v16bf {
            int kt = ktBase + st;
            return buildA(hrow, hrow[kt >> 2], (kt & 3) * 32, laneHi);
          };
          // software pipeline: B 2 ahead, A 1 ahead (tail over-reads are
          // unconsumed and land in allocated adjacent regions / padded LDS)
          v16bf b0 = loadB(0), b1 = loadB(1);
          v16bf a0 = mkA(0);
#pragma unroll 4
          for (int st = 0; st < 32; ++st) {
            __builtin_prefetch(Pb + (((size_t)(ktBase + st + 4) * 8 + n) * 32 + lane) * 16, 0, 0);
            v16bf b2 = loadB(st + 2);
            v16bf a1 = mkA(st + 1);
            c = wmma_bf16(a0, b0, c);
            a0 = a1; b0 = b1; b1 = b2;
          }
        }
        if (seg == 0) {                          // fold in  x_t @ Q^T  (K = 256)
          const float* xrow = &xstage[b][0];
          auto loadBQ = [&](int kt) -> v16bf {
            return *(const v16bf*)(Qb + (((size_t)kt * 8 + n) * 32 + lane) * 16);
          };
          v16bf b0 = loadBQ(0), b1 = loadBQ(1);
          v16bf a0 = buildA(xrow, 1.0f, 0, laneHi);
#pragma unroll
          for (int kt = 0; kt < 8; ++kt) {
            v16bf b2 = loadBQ(kt + 2);
            v16bf a1 = buildA(xrow, 1.0f, (kt + 1) * 32, laneHi);
            c = wmma_bf16(a0, b0, c);
            a0 = a1; b0 = b1; b1 = b2;
          }
        }
        float* pdst = part + ((size_t)tile * 16 + seg) * 256;   // [16][16] row-major
#pragma unroll
        for (int r = 0; r < 8; ++r)
          pdst[(r + laneHi) * 16 + laneLo] = c[r];
      }
    } else if (t >= 1) {
      // out(t-1) = h(t-1) @ R^T + x_{t-1} @ S^T
      const int job = (wg - 32) * 8 + wl;        // 0..31
      const int m = job >> 4, no = job & 15;
      const int b = m * 16 + laneLo;
      const float* hrow = &hstage[b][0];
      const float* xrow = &xstage[b][0];
      v8f c = {0.f, 0.f, 0.f, 0.f, 0.f, 0.f, 0.f, 0.f};
      {
        auto loadBR = [&](int kt) -> v16bf {
          return *(const v16bf*)(Rb + (((size_t)kt * 16 + no) * 32 + lane) * 16);
        };
        v16bf b0 = loadBR(0), b1 = loadBR(1);
        v16bf a0 = buildA(hrow, 1.0f, 0, laneHi);
#pragma unroll
        for (int kt = 0; kt < 4; ++kt) {         // K = 128 over R
          v16bf b2 = loadBR(kt + 2);
          v16bf a1 = buildA(hrow, 1.0f, (kt + 1) * 32, laneHi);
          c = wmma_bf16(a0, b0, c);
          a0 = a1; b0 = b1; b1 = b2;
        }
      }
      {
        auto loadBS = [&](int kt) -> v16bf {
          return *(const v16bf*)(Sb + (((size_t)kt * 16 + no) * 32 + lane) * 16);
        };
        v16bf b0 = loadBS(0), b1 = loadBS(1);
        v16bf a0 = buildA(xrow, 1.0f, 0, laneHi);
#pragma unroll
        for (int kt = 0; kt < 8; ++kt) {         // K = 256 over S
          v16bf b2 = loadBS(kt + 2);
          v16bf a1 = buildA(xrow, 1.0f, (kt + 1) * 32, laneHi);
          c = wmma_bf16(a0, b0, c);
          a0 = a1; b0 = b1; b1 = b2;
        }
      }
#pragma unroll
      for (int r = 0; r < 8; ++r) {
        int row = m * 16 + r + laneHi;
        outg[((size_t)row * S_ + (t - 1)) * 256 + no * 16 + laneLo] = c[r];
      }
    }

    epoch++; grid_sync(cnt, epoch * NWG);

    // ---- phase B: reduce K-partials (XQ already in seg0) + LayerNorm ----
    if (wg < 2 && t < S_) {
      const int mt = wg;
      const int r  = tid & 15;                   // row within m-tile
      const int cb = tid >> 4;                   // 16 col-blocks of 8
      float acc[8];
#pragma unroll
      for (int cc = 0; cc < 8; ++cc) acc[cc] = 0.f;
      for (int seg = 0; seg < 16; ++seg) {
#pragma unroll
        for (int cc = 0; cc < 8; ++cc) {
          int col = cb * 8 + cc;
          int n = col >> 4, cl = col & 15;
          acc[cc] += part[(((size_t)(mt * 8 + n)) * 16 + seg) * 256 + r * 16 + cl];
        }
      }
      float s = 0.f, s2 = 0.f;
#pragma unroll
      for (int cc = 0; cc < 8; ++cc) { s += acc[cc]; s2 += acc[cc] * acc[cc]; }
      red[0][r][cb] = s; red[1][r][cb] = s2;
      __syncthreads();
      float mu = 0.f, m2 = 0.f;
#pragma unroll
      for (int q = 0; q < 16; ++q) { mu += red[0][r][q]; m2 += red[1][r][q]; }
      mu *= (1.f / 128.f); m2 *= (1.f / 128.f);
      float rstd = rsqrtf(m2 - mu * mu + EPS_);
      float* hdst = hbuf + (t & 1) * (32 * 128) + (mt * 16 + r) * 128;
#pragma unroll
      for (int cc = 0; cc < 8; ++cc) {
        int col = cb * 8 + cc;
        hdst[col] = (acc[cc] - mu) * rstd * gamma[col] + beta[col];
      }
    }

    epoch++; grid_sync(cnt, epoch * NWG);
  }
}

// ---------------------------------------------------------------------------
extern "C" void kernel_launch(void* const* d_in, const int* in_sizes, int n_in,
                              void* d_out, int out_size, void* d_ws, size_t ws_size,
                              hipStream_t stream) {
  (void)in_sizes; (void)n_in; (void)out_size; (void)ws_size;
  const float* x     = (const float*)d_in[0];   // [32,4096,256]
  const float* P     = (const float*)d_in[1];   // [128,128,128]
  const float* Q     = (const float*)d_in[2];   // [128,256]
  const float* R     = (const float*)d_in[3];   // [256,128]
  const float* Smat  = (const float*)d_in[4];   // [256,256]
  const float* gamma = (const float*)d_in[5];   // [128]
  const float* beta  = (const float*)d_in[6];   // [128]
  char* ws = (char*)d_ws;

  hipMemsetAsync(ws + CNT_OFF, 0, 64, stream);  // grid-barrier counter must start at 0

  // Pflat is already [K=16384, N=128] flat: P[(i*128+j)*128 + k]
  pack_wt<<<512, 256, 0, stream>>>(P, (unsigned short*)(ws + PB_OFF), 128, 16384, 1);
  pack_wt<<<8,   256, 0, stream>>>(Q,    (unsigned short*)(ws + QB_OFF), 128, 256, 0);
  pack_wt<<<8,   256, 0, stream>>>(R,    (unsigned short*)(ws + RB_OFF), 256, 128, 0);
  pack_wt<<<16,  256, 0, stream>>>(Smat, (unsigned short*)(ws + SB_OFF), 256, 256, 0);

  rqs_persistent<<<NWG, NTHR, 0, stream>>>(x, gamma, beta, (float*)d_out, ws);
}
